// IJGNN2_43920335569130
// MI455X (gfx1250) — matrix operand order repaired and checked by reading, not compiled, not run on hardware
//
#include <hip/hip_runtime.h>
#include <hip/hip_bf16.h>
#include <math.h>

// ---------------------------------------------------------------------------
// Problem constants (from the reference)
// ---------------------------------------------------------------------------
#define N_NODES_C 50000
#define N_EDGES_C 800000
#define NF_C      32
#define EF_C      16
#define HNF_C     64
#define HEF_C     64
#define NF_OUT_C  32
#define EF_OUT_C  16
#define N_ITERS_C 4      // reference constant; device scalar not readable under graph capture

typedef __attribute__((ext_vector_type(16))) __bf16 v16bf;
typedef __attribute__((ext_vector_type(8)))  float  v8f;

// ---------------------------------------------------------------------------
// WMMA core: D(f32 16x16) += A(bf16 16x32) * B(bf16 32x16)
// ---------------------------------------------------------------------------
__device__ __forceinline__ v8f wmma_bf16(v16bf a, v16bf b, v8f c) {
  // (neg_a, A, neg_b, B, c_mod, C, reuse_a, reuse_b)
  return __builtin_amdgcn_wmma_f32_16x16x32_bf16(false, a, false, b, (short)0, c,
                                                 false, false);
}

__device__ __forceinline__ float sigmoidf_(float x) { return 1.0f / (1.0f + expf(-x)); }

// A-fragment (16x32 bf16) element->K map per ISA 7.12.2:
//   lanes 0-15: row M = lane, V0..V3 hold K=0..7, V4..V7 hold K=16..23
//   lanes16-31: same rows, +8 K offset
__device__ __forceinline__ v16bf load_A_lds(const __bf16* base, int row_stride,
                                            int lane, int k0) {
  const int row  = lane & 15;
  const int koff = (lane >> 4) << 3;
  const __bf16* p = base + row * row_stride + k0 + koff;
  v16bf a;
#pragma unroll
  for (int e = 0; e < 16; ++e) a[e] = p[(e < 8) ? e : (e + 8)];
  return a;
}

// Same A layout, gathered directly from a row-major f32 matrix in global.
__device__ __forceinline__ v16bf load_A_gf32(const float* base, int row_stride,
                                             int lane, int k0) {
  const int row  = lane & 15;
  const int koff = (lane >> 4) << 3;
  const float* p = base + row * row_stride + k0 + koff;
  v16bf a;
#pragma unroll
  for (int e = 0; e < 16; ++e) a[e] = (__bf16)p[(e < 8) ? e : (e + 8)];
  return a;
}

// B-fragments pre-packed fragment-linear: P[((s*ntiles + t)*32 + lane)*16 + e]
__device__ __forceinline__ v16bf load_B(const __bf16* P, int s, int t, int ntiles,
                                        int lane) {
  return *(const v16bf*)(P + ((((s * ntiles + t) << 5) + lane) << 4));
}

__device__ __forceinline__ void atomicMaxF(float* addr, float val) {
  unsigned* ua  = (unsigned*)addr;
  unsigned  old = __hip_atomic_load(ua, __ATOMIC_RELAXED, __HIP_MEMORY_SCOPE_AGENT);
  while (true) {
    if (__uint_as_float(old) >= val) break;
    unsigned assumed = old;
    old = atomicCAS(ua, assumed, __float_as_uint(val));
    if (old == assumed) break;
  }
}

// ---------------------------------------------------------------------------
// Weight packing: W[out_d][in_d] (f32) -> bf16 B-fragments, K padded to 32
// B layout (matches A): lane n=lane&15; lanes 0-15 K=0..7,16..23; lanes 16-31 K=8..15,24..31
// ---------------------------------------------------------------------------
__device__ void pack_one(const float* __restrict__ W, __bf16* __restrict__ P,
                         int out_d, int in_d, int g, int gs) {
  const int ksteps = (in_d + 31) >> 5;
  const int ntiles = (out_d + 15) >> 4;
  const int total  = ksteps * ntiles * 512;
  for (int i = g; i < total; i += gs) {
    const int e    = i & 15;
    const int lane = (i >> 4) & 31;
    const int ft   = i >> 9;
    const int t    = ft % ntiles;
    const int s    = ft / ntiles;
    const int n    = t * 16 + (lane & 15);
    const int k    = s * 32 + ((lane >> 4) << 3) + ((e < 8) ? e : (e + 8));
    float v        = (n < out_d && k < in_d) ? W[n * in_d + k] : 0.0f;
    P[i] = (__bf16)v;
  }
}

__global__ void pack_weights_kernel(
    const float* We, const float* Wn, const float* Wihn, const float* Whhn,
    const float* Wihe, const float* Whhe, const float* Wno, const float* Weo,
    __bf16* WeP, __bf16* WnP, __bf16* WihnP, __bf16* WhhnP,
    __bf16* WiheP, __bf16* WhheP, __bf16* WnoP, __bf16* WeoP) {
  const int g  = blockIdx.x * blockDim.x + threadIdx.x;
  const int gs = gridDim.x * blockDim.x;
  pack_one(We,   WeP,   HEF_C,     2 * NF_C + EF_C, g, gs);
  pack_one(Wn,   WnP,   HNF_C,     NF_C + HEF_C,    g, gs);
  pack_one(Wihn, WihnP, 3 * HNF_C, HNF_C,           g, gs);
  pack_one(Whhn, WhhnP, 3 * HNF_C, HNF_C,           g, gs);
  pack_one(Wihe, WiheP, 3 * HEF_C, HEF_C,           g, gs);
  pack_one(Whhe, WhheP, 3 * HEF_C, HEF_C,           g, gs);
  pack_one(Wno,  WnoP,  NF_OUT_C,  HNF_C,           g, gs);
  pack_one(Weo,  WeoP,  EF_OUT_C,  HEF_C,           g, gs);
}

// ---------------------------------------------------------------------------
// Init kernels
// ---------------------------------------------------------------------------
__global__ void zero_state_kernel(float* hnf, float* hef) {
  const long g  = (long)blockIdx.x * blockDim.x + threadIdx.x;
  const long gs = (long)gridDim.x * blockDim.x;
  for (long i = g; i < (long)N_NODES_C * HNF_C; i += gs) hnf[i] = 0.0f;
  for (long i = g; i < (long)N_EDGES_C * HEF_C; i += gs) hef[i] = 0.0f;
}

__global__ void init_attn_kernel(float* aggsum, float* denom, float* nodeMax) {
  const long g  = (long)blockIdx.x * blockDim.x + threadIdx.x;
  const long gs = (long)gridDim.x * blockDim.x;
  for (long i = g; i < (long)N_NODES_C * HEF_C; i += gs) aggsum[i] = 0.0f;
  for (long i = g; i < N_NODES_C; i += gs) {
    denom[i]   = 0.0f;
    nodeMax[i] = -3.402823466e38f;
  }
}

// ---------------------------------------------------------------------------
// K1: edge update.  uef = relu([nf[src],nf[dst],ef] @ We^T + be)  (K=80->96)
//     logit = uef . Wa + ba ; atomic segment-max per dst
// One wave per 16 edges; 12 WMMAs; logit via in-wave shuffle reduction.
// ---------------------------------------------------------------------------
__global__ void __launch_bounds__(128)
edge_update_kernel(const float* __restrict__ nf, const float* __restrict__ ef,
                   const int* __restrict__ src, const int* __restrict__ dst,
                   const __bf16* __restrict__ WeP, const float* __restrict__ be,
                   const float* __restrict__ Wa, const float* __restrict__ ba,
                   float* __restrict__ uef, float* __restrict__ logit,
                   float* __restrict__ nodeMax) {
  __shared__ __bf16 sEM[4][16][96];
  const int wave = threadIdx.x >> 5;
  const int lane = threadIdx.x & 31;
  const int tile = blockIdx.x * 4 + wave;
  if (tile * 16 >= N_EDGES_C) return;  // wave-uniform
  const int e0 = tile * 16;

  __builtin_prefetch(WeP, 0, 1);  // global_prefetch of packed weights

  if (lane < 16) {
    const int eidx = e0 + lane;
    const int sN = src[eidx], dN = dst[eidx];
#pragma unroll
    for (int j = 0; j < NF_C; ++j) sEM[wave][lane][j] = (__bf16)nf[sN * NF_C + j];
#pragma unroll
    for (int j = 0; j < NF_C; ++j) sEM[wave][lane][NF_C + j] = (__bf16)nf[dN * NF_C + j];
#pragma unroll
    for (int j = 0; j < EF_C; ++j) sEM[wave][lane][2 * NF_C + j] = (__bf16)ef[eidx * EF_C + j];
#pragma unroll
    for (int j = 80; j < 96; ++j) sEM[wave][lane][j] = (__bf16)0.0f;  // K pad
  }
  asm volatile("s_wait_dscnt 0" ::: "memory");  // wave-private LDS handoff

  const int col = lane & 15, hl = lane >> 4;
  v8f acc[4];
#pragma unroll
  for (int t = 0; t < 4; ++t) {
    const float b = be[t * 16 + col];
#pragma unroll
    for (int v = 0; v < 8; ++v) acc[t][v] = b;
  }
#pragma unroll
  for (int s = 0; s < 3; ++s) {
    v16bf a = load_A_lds(&sEM[wave][0][0], 96, lane, s * 32);
#pragma unroll
    for (int t = 0; t < 4; ++t) acc[t] = wmma_bf16(a, load_B(WeP, s, t, 4, lane), acc[t]);
  }

  float part[8] = {0, 0, 0, 0, 0, 0, 0, 0};
#pragma unroll
  for (int t = 0; t < 4; ++t) {
    const float wa = Wa[t * 16 + col];
#pragma unroll
    for (int v = 0; v < 8; ++v) {
      const float u = fmaxf(acc[t][v], 0.0f);
      uef[(e0 + v + 8 * hl) * HEF_C + t * 16 + col] = u;
      part[v] += u * wa;
    }
  }
  // reduce over the 16 lanes that share the same row group (stay within halves)
#pragma unroll
  for (int m = 1; m < 16; m <<= 1) {
#pragma unroll
    for (int v = 0; v < 8; ++v) part[v] += __shfl_xor(part[v], m, 32);
  }
  if (col == 0) {  // lanes 0 and 16 own rows 0-7 / 8-15
#pragma unroll
    for (int v = 0; v < 8; ++v) {
      const int eidx = e0 + v + 8 * hl;
      const float lg = part[v] + ba[0];
      logit[eidx] = lg;
      atomicMaxF(&nodeMax[dst[eidx]], lg);
    }
  }
}

// ---------------------------------------------------------------------------
// K2: softmax accumulate. One thread per (edge, channel).
// ---------------------------------------------------------------------------
__global__ void attn_accum_kernel(const int* __restrict__ dst,
                                  const float* __restrict__ logit,
                                  const float* __restrict__ nodeMax,
                                  const float* __restrict__ uef,
                                  float* __restrict__ denom,
                                  float* __restrict__ aggsum) {
  const long g = (long)blockIdx.x * blockDim.x + threadIdx.x;
  if (g >= (long)N_EDGES_C * HEF_C) return;
  const int e = (int)(g >> 6);
  const int c = (int)(g & 63);
  const int d = dst[e];
  const float w = expf(logit[e] - nodeMax[d]);
  if (c == 0) atomicAdd(&denom[d], w);
  atomicAdd(&aggsum[d * HEF_C + c], w * uef[e * HEF_C + c]);
}

// ---------------------------------------------------------------------------
// Shared GRU tile: 16 rows, hidden 64, 3*64 gates.  48 WMMAs per wave.
// C-fragment slot (lane,v) carries the SAME channel n for gi/gh/r/z/c tiles,
// so all gate math is pure per-register VALU.
// ---------------------------------------------------------------------------
__device__ __forceinline__ void gru_tile(v16bf aU0, v16bf aU1, v16bf aH0, v16bf aH1,
                                         const __bf16* __restrict__ WihP,
                                         const float* __restrict__ bih,
                                         const __bf16* __restrict__ WhhP,
                                         const float* __restrict__ bhh,
                                         float* __restrict__ h, int row0, int lane) {
  const int col = lane & 15, hl = lane >> 4;
#pragma unroll
  for (int t = 0; t < 4; ++t) {
    v8f gir, giz, gic, ghr, ghz, ghc;
    const float bir = bih[t * 16 + col], bhr = bhh[t * 16 + col];
    const float biz = bih[(t + 4) * 16 + col], bhz = bhh[(t + 4) * 16 + col];
    const float bic = bih[(t + 8) * 16 + col], bhc = bhh[(t + 8) * 16 + col];
#pragma unroll
    for (int v = 0; v < 8; ++v) {
      gir[v] = bir; giz[v] = biz; gic[v] = bic;
      ghr[v] = bhr; ghz[v] = bhz; ghc[v] = bhc;
    }
    gir = wmma_bf16(aU0, load_B(WihP, 0, t,      12, lane), gir);
    gir = wmma_bf16(aU1, load_B(WihP, 1, t,      12, lane), gir);
    giz = wmma_bf16(aU0, load_B(WihP, 0, t + 4,  12, lane), giz);
    giz = wmma_bf16(aU1, load_B(WihP, 1, t + 4,  12, lane), giz);
    gic = wmma_bf16(aU0, load_B(WihP, 0, t + 8,  12, lane), gic);
    gic = wmma_bf16(aU1, load_B(WihP, 1, t + 8,  12, lane), gic);
    ghr = wmma_bf16(aH0, load_B(WhhP, 0, t,      12, lane), ghr);
    ghr = wmma_bf16(aH1, load_B(WhhP, 1, t,      12, lane), ghr);
    ghz = wmma_bf16(aH0, load_B(WhhP, 0, t + 4,  12, lane), ghz);
    ghz = wmma_bf16(aH1, load_B(WhhP, 1, t + 4,  12, lane), ghz);
    ghc = wmma_bf16(aH0, load_B(WhhP, 0, t + 8,  12, lane), ghc);
    ghc = wmma_bf16(aH1, load_B(WhhP, 1, t + 8,  12, lane), ghc);
#pragma unroll
    for (int v = 0; v < 8; ++v) {
      const int   ridx = (row0 + v + 8 * hl) * 64 + t * 16 + col;
      const float r    = sigmoidf_(gir[v] + ghr[v]);
      const float z    = sigmoidf_(giz[v] + ghz[v]);
      const float cg   = tanhf(gic[v] + r * ghc[v]);
      const float hp   = h[ridx];
      h[ridx]          = (1.0f - z) * cg + z * hp;
    }
  }
}

// ---------------------------------------------------------------------------
// K3: node update + node GRU.  unf = relu([nf, agg] @ Wn^T + bn); hnf = GRU(unf,hnf)
// One wave per 16 nodes; 12 + 48 WMMAs.
// ---------------------------------------------------------------------------
__global__ void __launch_bounds__(128)
node_update_kernel(const float* __restrict__ nf, const float* __restrict__ aggsum,
                   const float* __restrict__ denom,
                   const __bf16* __restrict__ WnP, const float* __restrict__ bn,
                   const __bf16* __restrict__ WihP, const float* __restrict__ bih,
                   const __bf16* __restrict__ WhhP, const float* __restrict__ bhh,
                   float* __restrict__ hnf) {
  __shared__ __bf16 sX[4][16][96];
  __shared__ __bf16 sU[4][16][64];
  const int wave = threadIdx.x >> 5;
  const int lane = threadIdx.x & 31;
  const int tile = blockIdx.x * 4 + wave;
  if (tile * 16 >= N_NODES_C) return;  // wave-uniform (no block barrier used)
  const int n0 = tile * 16;

  if (lane < 16) {
    const int nid = n0 + lane;
#pragma unroll
    for (int j = 0; j < NF_C; ++j) sX[wave][lane][j] = (__bf16)nf[nid * NF_C + j];
    const float dn = denom[nid] + 1e-16f;
#pragma unroll
    for (int j = 0; j < HEF_C; ++j)
      sX[wave][lane][NF_C + j] = (__bf16)(aggsum[nid * HEF_C + j] / dn);
  }
  asm volatile("s_wait_dscnt 0" ::: "memory");

  const int col = lane & 15, hl = lane >> 4;
  v8f u[4];
#pragma unroll
  for (int t = 0; t < 4; ++t) {
    const float b = bn[t * 16 + col];
#pragma unroll
    for (int v = 0; v < 8; ++v) u[t][v] = b;
  }
#pragma unroll
  for (int s = 0; s < 3; ++s) {
    v16bf a = load_A_lds(&sX[wave][0][0], 96, lane, s * 32);
#pragma unroll
    for (int t = 0; t < 4; ++t) u[t] = wmma_bf16(a, load_B(WnP, s, t, 4, lane), u[t]);
  }
  // C-frag (rows in vgprs) -> A-frag (rows in lanes): stage relu(unf) via LDS
#pragma unroll
  for (int t = 0; t < 4; ++t)
#pragma unroll
    for (int v = 0; v < 8; ++v)
      sU[wave][v + 8 * hl][t * 16 + col] = (__bf16)fmaxf(u[t][v], 0.0f);
  asm volatile("s_wait_dscnt 0" ::: "memory");

  v16bf aU0 = load_A_lds(&sU[wave][0][0], 64, lane, 0);
  v16bf aU1 = load_A_lds(&sU[wave][0][0], 64, lane, 32);
  v16bf aH0 = load_A_gf32(hnf + (long)n0 * HNF_C, HNF_C, lane, 0);
  v16bf aH1 = load_A_gf32(hnf + (long)n0 * HNF_C, HNF_C, lane, 32);
  gru_tile(aU0, aU1, aH0, aH1, WihP, bih, WhhP, bhh, hnf, n0, lane);
}

// ---------------------------------------------------------------------------
// K4: edge GRU.  hef = GRU(uef, hef).  One wave per 16 edges; 48 WMMAs.
// ---------------------------------------------------------------------------
__global__ void __launch_bounds__(128)
edge_gru_kernel(const float* __restrict__ uef,
                const __bf16* __restrict__ WihP, const float* __restrict__ bih,
                const __bf16* __restrict__ WhhP, const float* __restrict__ bhh,
                float* __restrict__ hef) {
  const int wave = threadIdx.x >> 5;
  const int lane = threadIdx.x & 31;
  const int tile = blockIdx.x * 4 + wave;
  if (tile * 16 >= N_EDGES_C) return;
  const int e0 = tile * 16;
  __builtin_prefetch(WihP, 0, 1);

  v16bf aU0 = load_A_gf32(uef + (long)e0 * HEF_C, HEF_C, lane, 0);
  v16bf aU1 = load_A_gf32(uef + (long)e0 * HEF_C, HEF_C, lane, 32);
  v16bf aH0 = load_A_gf32(hef + (long)e0 * HEF_C, HEF_C, lane, 0);
  v16bf aH1 = load_A_gf32(hef + (long)e0 * HEF_C, HEF_C, lane, 32);
  gru_tile(aU0, aU1, aH0, aH1, WihP, bih, WhhP, bhh, hef, e0, lane);
}

// ---------------------------------------------------------------------------
// K5/K6: output projections
// ---------------------------------------------------------------------------
__global__ void __launch_bounds__(128)
out_node_kernel(const float* __restrict__ hnf, const __bf16* __restrict__ WnoP,
                const float* __restrict__ bno, float* __restrict__ out) {
  const int wave = threadIdx.x >> 5;
  const int lane = threadIdx.x & 31;
  const int tile = blockIdx.x * 4 + wave;
  if (tile * 16 >= N_NODES_C) return;
  const int n0 = tile * 16;
  const int col = lane & 15, hl = lane >> 4;
  v16bf a0 = load_A_gf32(hnf + (long)n0 * HNF_C, HNF_C, lane, 0);
  v16bf a1 = load_A_gf32(hnf + (long)n0 * HNF_C, HNF_C, lane, 32);
#pragma unroll
  for (int t = 0; t < 2; ++t) {
    v8f c;
    const float b = bno[t * 16 + col];
#pragma unroll
    for (int v = 0; v < 8; ++v) c[v] = b;
    c = wmma_bf16(a0, load_B(WnoP, 0, t, 2, lane), c);
    c = wmma_bf16(a1, load_B(WnoP, 1, t, 2, lane), c);
#pragma unroll
    for (int v = 0; v < 8; ++v)
      out[(n0 + v + 8 * hl) * NF_OUT_C + t * 16 + col] = c[v];
  }
}

__global__ void __launch_bounds__(128)
out_edge_kernel(const float* __restrict__ hef, const __bf16* __restrict__ WeoP,
                const float* __restrict__ beo, float* __restrict__ out) {
  const int wave = threadIdx.x >> 5;
  const int lane = threadIdx.x & 31;
  const int tile = blockIdx.x * 4 + wave;
  if (tile * 16 >= N_EDGES_C) return;
  const int e0 = tile * 16;
  const int col = lane & 15, hl = lane >> 4;
  v16bf a0 = load_A_gf32(hef + (long)e0 * HEF_C, HEF_C, lane, 0);
  v16bf a1 = load_A_gf32(hef + (long)e0 * HEF_C, HEF_C, lane, 32);
  v8f c;
  const float b = beo[col];
#pragma unroll
  for (int v = 0; v < 8; ++v) c[v] = b;
  c = wmma_bf16(a0, load_B(WeoP, 0, 0, 1, lane), c);
  c = wmma_bf16(a1, load_B(WeoP, 1, 0, 1, lane), c);
#pragma unroll
  for (int v = 0; v < 8; ++v)
    out[(e0 + v + 8 * hl) * EF_OUT_C + col] = c[v];
}

// ---------------------------------------------------------------------------
// Host launcher
// ---------------------------------------------------------------------------
extern "C" void kernel_launch(void* const* d_in, const int* in_sizes, int n_in,
                              void* d_out, int out_size, void* d_ws, size_t ws_size,
                              hipStream_t stream) {
  (void)in_sizes; (void)n_in; (void)out_size; (void)ws_size;
  const float* nf   = (const float*)d_in[0];
  const float* ef   = (const float*)d_in[1];
  const int*   src  = (const int*)d_in[2];
  const int*   dst  = (const int*)d_in[3];
  // d_in[4] = n_iters (reference constant == 4)
  const float* We   = (const float*)d_in[5];
  const float* be   = (const float*)d_in[6];
  const float* Wa   = (const float*)d_in[7];
  const float* ba   = (const float*)d_in[8];
  const float* Wn   = (const float*)d_in[9];
  const float* bn   = (const float*)d_in[10];
  const float* Wihn = (const float*)d_in[11];
  const float* bihn = (const float*)d_in[12];
  const float* Whhn = (const float*)d_in[13];
  const float* bhhn = (const float*)d_in[14];
  const float* Wihe = (const float*)d_in[15];
  const float* bihe = (const float*)d_in[16];
  const float* Whhe = (const float*)d_in[17];
  const float* bhhe = (const float*)d_in[18];
  const float* Wno  = (const float*)d_in[19];
  const float* bno  = (const float*)d_in[20];
  const float* Weo  = (const float*)d_in[21];
  const float* beo  = (const float*)d_in[22];

  char* wsb = (char*)d_ws;
  size_t off = 0;
  auto take = [&](size_t bytes) -> char* {
    char* p = wsb + off;
    off = (off + bytes + 255) & ~(size_t)255;
    return p;
  };
  float*  uef     = (float*)take((size_t)N_EDGES_C * HEF_C * 4);
  float*  hef     = (float*)take((size_t)N_EDGES_C * HEF_C * 4);
  float*  hnf     = (float*)take((size_t)N_NODES_C * HNF_C * 4);
  float*  aggsum  = (float*)take((size_t)N_NODES_C * HEF_C * 4);
  float*  denom   = (float*)take((size_t)N_NODES_C * 4);
  float*  nodeMax = (float*)take((size_t)N_NODES_C * 4);
  float*  logit   = (float*)take((size_t)N_EDGES_C * 4);
  __bf16* WeP     = (__bf16*)take(3 * 4 * 512 * 2);
  __bf16* WnP     = (__bf16*)take(3 * 4 * 512 * 2);
  __bf16* WihnP   = (__bf16*)take(2 * 12 * 512 * 2);
  __bf16* WhhnP   = (__bf16*)take(2 * 12 * 512 * 2);
  __bf16* WiheP   = (__bf16*)take(2 * 12 * 512 * 2);
  __bf16* WhheP   = (__bf16*)take(2 * 12 * 512 * 2);
  __bf16* WnoP    = (__bf16*)take(2 * 2 * 512 * 2);
  __bf16* WeoP    = (__bf16*)take(2 * 1 * 512 * 2);

  pack_weights_kernel<<<64, 256, 0, stream>>>(We, Wn, Wihn, Whhn, Wihe, Whhe, Wno, Weo,
                                              WeP, WnP, WihnP, WhhnP, WiheP, WhheP,
                                              WnoP, WeoP);
  zero_state_kernel<<<4096, 256, 0, stream>>>(hnf, hef);

  const int EDGE_BLOCKS = (N_EDGES_C / 16 + 3) / 4;  // 12500
  const int NODE_BLOCKS = (N_NODES_C / 16 + 3) / 4;  // 782
  for (int it = 0; it < N_ITERS_C; ++it) {
    init_attn_kernel<<<4096, 256, 0, stream>>>(aggsum, denom, nodeMax);
    edge_update_kernel<<<EDGE_BLOCKS, 128, 0, stream>>>(nf, ef, src, dst, WeP, be, Wa,
                                                        ba, uef, logit, nodeMax);
    attn_accum_kernel<<<(N_EDGES_C * 64) / 256, 256, 0, stream>>>(dst, logit, nodeMax,
                                                                  uef, denom, aggsum);
    node_update_kernel<<<NODE_BLOCKS, 128, 0, stream>>>(nf, aggsum, denom, WnP, bn,
                                                        WihnP, bihn, WhhnP, bhhn, hnf);
    edge_gru_kernel<<<EDGE_BLOCKS, 128, 0, stream>>>(uef, WiheP, bihe, WhheP, bhhe, hef);
  }

  float* out_nf = (float*)d_out;
  float* out_ef = out_nf + (size_t)N_NODES_C * NF_OUT_C;
  out_node_kernel<<<NODE_BLOCKS, 128, 0, stream>>>(hnf, WnoP, bno, out_nf);
  out_edge_kernel<<<EDGE_BLOCKS, 128, 0, stream>>>(hef, WeoP, beo, out_ef);
}
